// LinearHeadScaledCosNorm_80960133530133
// MI455X (gfx1250) — matrix operand order repaired and checked by reading, not compiled, not run on hardware
//
#include <hip/hip_runtime.h>

typedef __attribute__((ext_vector_type(16))) _Float16 v16h;
typedef __attribute__((ext_vector_type(8)))  _Float16 v8h;
typedef __attribute__((ext_vector_type(4))) _Float16 v4h;
typedef __attribute__((ext_vector_type(8)))  float    v8f;
typedef __attribute__((ext_vector_type(4)))  float    v4f;

#define KDIM 768
#define OUTD 100
#define OPAD 112            // 7 tiles of 16 (rows 100..111 zero)
#define LDSS 776            // LDS row stride in halfs: 768+8 -> conflict-free ds_load_b128
#define NTHREADS 512        // 16 waves -> 4 waves/SIMD for latency hiding
#define ROWS_PER_WG 256     // 16 waves x 16 x-rows
#define KSTEPS (KDIM / 32)  // 24

// ---- Phase 1: W (100x768 f32) -> Wh (112x768 f16, padded rows zeroed) ----
__global__ void prep_w_kernel(const float* __restrict__ W, _Float16* __restrict__ Wh) {
    int idx = blockIdx.x * blockDim.x + threadIdx.x;
    if (idx >= OPAD * KDIM) return;
    int o = idx / KDIM;
    int k = idx - o * KDIM;
    float v = (o < OUTD) ? W[o * KDIM + k] : 0.0f;
    Wh[idx] = (_Float16)v;
}

// A-fragment LDS address for (ks, ct); layout: 16-bit A 16x32,
// lanes 0-15 K 0-7 & 16-23, lanes 16-31 K 8-15 & 24-31.
__device__ __forceinline__ const _Float16*
afrag_addr(const _Float16* sW, int m, int sel, int ks, int ct) {
    return &sW[(ct * 16 + m) * LDSS + ks * 32 + sel * 8];
}

// ---- Phase 2: flipped-GEMM streaming kernel ----
// A = W tile (16 out-cols x 32 K) from LDS (rotated 2 fragments ahead),
// B = x^T tile (32 K x 16 x-rows) streamed NT from HBM via register rotation.
// Per-row L2 norm folded into the fp32 stream; applied at the output.
__global__ __launch_bounds__(NTHREADS)
void cosnorm_head_kernel(const float* __restrict__ x,
                         const _Float16* __restrict__ Wh,
                         const float* __restrict__ scale,
                         float* __restrict__ out) {
    extern __shared__ char smem_raw[];
    _Float16* sW = (_Float16*)smem_raw;            // OPAD * LDSS halfs (~174 KB)

    const int tid = threadIdx.x;

    // ---- one-time: stage f16 weights into LDS (regular temporal: keep in L2) ----
    {
        const int nvec = OPAD * (KDIM / 8);        // 10752 x v8h
        for (int j = tid; j < nvec; j += NTHREADS) {
            int row = j / (KDIM / 8);
            int c   = (j - row * (KDIM / 8)) * 8;
            v8h v = *(const v8h*)(Wh + row * KDIM + c);
            *(v8h*)(&sW[row * LDSS + c]) = v;
        }
    }
    __syncthreads();

    const int wave = tid >> 5;
    const int lane = tid & 31;
    const int m    = lane & 15;   // A: out-col in tile; B: x-row in tile
    const int sel  = lane >> 4;   // K-half selector

    // This lane's x row (the B-matrix column). B layout (32x16, 16-bit):
    // lanes 0-15 hold K 0-7 & 16-23, lanes 16-31 hold K 8-15 & 24-31 of each k-step.
    const size_t xrow = (size_t)blockIdx.x * ROWS_PER_WG + wave * 16 + m;
    const float* xp = x + xrow * KDIM + sel * 8;

    v8f acc[7];
    #pragma unroll
    for (int ct = 0; ct < 7; ++ct) acc[ct] = (v8f){0,0,0,0,0,0,0,0};

    float ss = 0.0f;

    // register-rotated NT prefetch of this lane's 16 fp32 x values (two runs of 8, 16 apart)
    v4f c0 = __builtin_nontemporal_load((const v4f*)(xp +  0));
    v4f c1 = __builtin_nontemporal_load((const v4f*)(xp +  4));
    v4f c2 = __builtin_nontemporal_load((const v4f*)(xp + 16));
    v4f c3 = __builtin_nontemporal_load((const v4f*)(xp + 20));

    // A-fragment 2-deep rotation: preload fragments (0,0) and (0,1)
    v8h a0, a1, n0, n1;
    {
        const _Float16* p0 = afrag_addr(sW, m, sel, 0, 0);
        a0 = *(const v8h*)(p0);
        a1 = *(const v8h*)(p0 + 16);
        const _Float16* p1 = afrag_addr(sW, m, sel, 0, 1);
        n0 = *(const v8h*)(p1);
        n1 = *(const v8h*)(p1 + 16);
    }

    for (int ks = 0; ks < KSTEPS; ++ks) {
        v4f f0 = c0, f1 = c1, f2 = c2, f3 = c3;

        // prefetch next k-step's x while this one computes (clamped last load)
        const int ksn = (ks + 1 < KSTEPS) ? (ks + 1) : ks;
        const float* np = xp + ksn * 32;
        c0 = __builtin_nontemporal_load((const v4f*)(np +  0));
        c1 = __builtin_nontemporal_load((const v4f*)(np +  4));
        c2 = __builtin_nontemporal_load((const v4f*)(np + 16));
        c3 = __builtin_nontemporal_load((const v4f*)(np + 20));

        // fold sum-of-squares (fp32, pre-rounding) into the stream
        #pragma unroll
        for (int e = 0; e < 4; ++e) {
            ss = fmaf(f0[e], f0[e], ss);
            ss = fmaf(f1[e], f1[e], ss);
            ss = fmaf(f2[e], f2[e], ss);
            ss = fmaf(f3[e], f3[e], ss);
        }

        // convert to the f16 B fragment
        v4h h0 = __builtin_convertvector(f0, v4h);
        v4h h1 = __builtin_convertvector(f1, v4h);
        v4h h2 = __builtin_convertvector(f2, v4h);
        v4h h3 = __builtin_convertvector(f3, v4h);
        v8h blo = __builtin_shufflevector(h0, h1, 0,1,2,3,4,5,6,7);
        v8h bhi = __builtin_shufflevector(h2, h3, 0,1,2,3,4,5,6,7);
        v16h B  = __builtin_shufflevector(blo, bhi, 0,1,2,3,4,5,6,7,8,9,10,11,12,13,14,15);

        // 7 column tiles; A fragments rotated two LDS load-pairs ahead so each
        // WMMA's operands landed two WMMA-durations ago
        #pragma unroll
        for (int ct = 0; ct < 7; ++ct) {
            v16h A = __builtin_shufflevector(a0, a1, 0,1,2,3,4,5,6,7,8,9,10,11,12,13,14,15);

            // prefetch fragment (ks, ct+2) with wraparound into the next k-step
            const int i2  = ct + 2;
            int nks = ks + (i2 >= 7 ? 1 : 0);
            const int nct = (i2 >= 7) ? (i2 - 7) : i2;
            if (nks >= KSTEPS) nks = KSTEPS - 1;   // clamped redundant prefetch
            const _Float16* ap = afrag_addr(sW, m, sel, nks, nct);
            v8h p0 = *(const v8h*)(ap);
            v8h p1 = *(const v8h*)(ap + 16);

            acc[ct] = __builtin_amdgcn_wmma_f32_16x16x32_f16(
                /*neg_a=*/false, A, /*neg_b=*/false, B,
                /*c_mod=*/(short)0, acc[ct], /*reuse_a=*/false, /*reuse_b=*/false);

            a0 = n0; a1 = n1;
            n0 = p0; n1 = p1;
        }
    }

    // finish the per-row norm: lanes n and n+16 hold the two K-halves
    ss += __shfl_xor(ss, 16);
    const float rn = scale[0] / fmaxf(sqrtf(ss), 1e-12f);

    // store: lane's out row = xrow; accumulator VGPR v -> out col ct*16+sel*8+v
    // => consecutive floats => pure NT b128 stores (row stride 400 B, 16B-aligned)
    float* op = out + xrow * OUTD;
    #pragma unroll
    for (int ct = 0; ct < 7; ++ct) {
        v4f lo = { acc[ct][0] * rn, acc[ct][1] * rn, acc[ct][2] * rn, acc[ct][3] * rn };
        v4f hi = { acc[ct][4] * rn, acc[ct][5] * rn, acc[ct][6] * rn, acc[ct][7] * rn };
        const int o0 = ct * 16 + sel * 8;
        if (ct < 6) {
            __builtin_nontemporal_store(lo, (v4f*)(op + o0));
            __builtin_nontemporal_store(hi, (v4f*)(op + o0 + 4));
        } else if (sel == 0) {
            __builtin_nontemporal_store(lo, (v4f*)(op + 96));  // cols 96..99
        }
    }
}

extern "C" void kernel_launch(void* const* d_in, const int* in_sizes, int n_in,
                              void* d_out, int out_size, void* d_ws, size_t ws_size,
                              hipStream_t stream) {
    const float* x     = (const float*)d_in[0];
    const float* W     = (const float*)d_in[1];
    const float* scale = (const float*)d_in[2];
    float*       out   = (float*)d_out;
    _Float16*    Wh    = (_Float16*)d_ws;   // 112*768*2 = 172 KB scratch

    {
        int total  = OPAD * KDIM;
        int blocks = (total + 255) / 256;
        prep_w_kernel<<<blocks, 256, 0, stream>>>(W, Wh);
    }
    {
        int n      = in_sizes[0] / KDIM;          // 262144
        int blocks = n / ROWS_PER_WG;             // 1024
        size_t lds = (size_t)OPAD * LDSS * 2;     // 173,824 B
        cosnorm_head_kernel<<<blocks, NTHREADS, lds, stream>>>(x, Wh, scale, out);
    }
}
